// PixelPrototypeClassifier_21449066676524
// MI455X (gfx1250) — matrix operand order repaired and checked by reading, not compiled, not run on hardware
//
#include <hip/hip_runtime.h>

// ---------------------------------------------------------------------------
// PixelPrototypeClassifier for MI455X (gfx1250), wave32 + WMMA.
// ---------------------------------------------------------------------------
typedef __attribute__((ext_vector_type(16))) _Float16 v16h;
typedef __attribute__((ext_vector_type(8)))  _Float16 v8h;
typedef __attribute__((ext_vector_type(8)))  float    v8f;

#define N_PIX   73728      // 8*96*96
#define FEAT    512
#define K_CLS   19
#define M_PRO   10
#define KM      190
#define KM_PAD  192
#define HWP     9216       // 96*96
#define NCHUNK  36         // N_PIX / 2048
#define BNBLK   288        // N_PIX / 256

// output layout (floats): out_seg | proto_logits | proto_target | new_protos
#define OUT_SEG_OFF 0
#define OUT_LOG_OFF 1400832u     // 8*19*9216
#define OUT_TGT_OFF 15409152u    // + 73728*190
#define OUT_PRO_OFF 15482880u    // + 73728

__device__ __forceinline__ float safef(float x) { return x == 0.0f ? 1.0f : x; }

template <int BS>
__device__ __forceinline__ float breduce(float v, float* red) {
  int t = threadIdx.x;
  red[t] = v;
  __syncthreads();
#pragma unroll
  for (int s = BS / 2; s > 0; s >>= 1) {
    if (t < s) red[t] += red[t + s];
    __syncthreads();
  }
  float r = red[0];
  __syncthreads();
  return r;
}

// A fragment: 16x32 f16 (M x K), rows row-major with leading dim ld.
// ISA 7.12.2: lane L(0..15)=row L holds K{0..7,16..23}; lane L+16 holds K{8..15,24..31}.
__device__ __forceinline__ v16h load_fragA(const _Float16* __restrict__ base, int ld) {
  int lane = threadIdx.x & 31;
  int r = lane & 15;
  int hi = lane >> 4;
  const _Float16* p = base + (size_t)r * ld + (hi << 3);
  v8h a = *(const v8h*)p;
  v8h b = *(const v8h*)(p + 16);
  v16h o;
#pragma unroll
  for (int i = 0; i < 8; ++i) { o[i] = a[i]; o[i + 8] = b[i]; }
  return o;
}

// B fragment: 32x16 f16 (K x N); source is row-major weight rows W[n][k] (math B = W^T).
// Same k order as A: lane n(0..15)=col n holds K{0..7,16..23};
// lane n+16 holds K{8..15,24..31} -> two 16B loads per lane.
__device__ __forceinline__ v16h load_fragB(const _Float16* __restrict__ base, int ld) {
  int lane = threadIdx.x & 31;
  int n = lane & 15;
  int hi = lane >> 4;
  const _Float16* p = base + (size_t)n * ld + (hi << 3);
  v8h a = *(const v8h*)p;
  v8h b = *(const v8h*)(p + 16);
  v16h o;
#pragma unroll
  for (int i = 0; i < 8; ++i) { o[i] = a[i]; o[i + 8] = b[i]; }
  return o;
}

// ---------------------------------------------------------------------------
// 1) proj_w f32 -> f16
__global__ __launch_bounds__(256) void k_convw(const float* __restrict__ w, _Float16* __restrict__ w16) {
  int i = blockIdx.x * 256 + threadIdx.x;
  if (i < FEAT * FEAT) w16[i] = (_Float16)w[i];
}

// 2) prototypes -> l2-normalized Pn (f32 [190,512], f16 [192,512] zero padded)
__global__ __launch_bounds__(256) void k_protos(const float* __restrict__ p, float* __restrict__ pn32,
                                                _Float16* __restrict__ pn16) {
  __shared__ float red[256];
  int r = blockIdx.x;
  int t = threadIdx.x;
  if (r >= KM) {  // zero pad rows 190,191 (f16 only)
    pn16[(size_t)r * FEAT + t] = (_Float16)0.0f;
    pn16[(size_t)r * FEAT + t + 256] = (_Float16)0.0f;
    return;
  }
  float a = p[(size_t)r * FEAT + t];
  float b = p[(size_t)r * FEAT + t + 256];
  float nrm = sqrtf(breduce<256>(a * a + b * b, red)) + 1e-10f;
  a /= nrm; b /= nrm;
  pn32[(size_t)r * FEAT + t] = a;
  pn32[(size_t)r * FEAT + t + 256] = b;
  pn16[(size_t)r * FEAT + t] = (_Float16)a;
  pn16[(size_t)r * FEAT + t + 256] = (_Float16)b;
}

// 3) features [8,512,96,96] f32 -> flat16 [N,512] f16 (transpose C<->HW, convert)
__global__ void k_transpose(const float* __restrict__ F, _Float16* __restrict__ X) {
  __shared__ float tile[32][33];
  int p0 = blockIdx.x * 32, c0 = blockIdx.y * 32, b = blockIdx.z;
  int tx = threadIdx.x, ty = threadIdx.y;
#pragma unroll
  for (int i = 0; i < 4; ++i) {
    int ch = c0 + ty + i * 8;
    tile[ty + i * 8][tx] = F[((size_t)(b * FEAT + ch)) * HWP + p0 + tx];
  }
  __syncthreads();
#pragma unroll
  for (int i = 0; i < 4; ++i) {
    int pl = ty + i * 8;
    X[((size_t)(b * HWP + p0 + pl)) * FEAT + c0 + tx] = (_Float16)tile[tx][pl];
  }
}

// 4) GEMM1: H[N,512] = X[N,512] @ W[512,512]^T + bias.  64x64 block tile, 8 waves.
__global__ __launch_bounds__(256) void k_gemm1(const _Float16* __restrict__ X, const _Float16* __restrict__ W,
                                               const float* __restrict__ bias, float* __restrict__ H) {
  int w = threadIdx.x >> 5;
  int lane = threadIdx.x & 31;
  int wm = w >> 2;  // 0..1
  int wn = w & 3;   // 0..3
  int row0 = blockIdx.y * 64 + wm * 32;
  int col0 = blockIdx.x * 64 + wn * 16;
  v8f acc0 = {}, acc1 = {};
  for (int kk = 0; kk < FEAT; kk += 32) {
    v16h a0 = load_fragA(X + (size_t)row0 * FEAT + kk, FEAT);
    v16h a1 = load_fragA(X + (size_t)(row0 + 16) * FEAT + kk, FEAT);
    v16h bb = load_fragB(W + (size_t)col0 * FEAT + kk, FEAT);
    acc0 = __builtin_amdgcn_wmma_f32_16x16x32_f16(false, a0, false, bb, (short)0, acc0, false, false);
    acc1 = __builtin_amdgcn_wmma_f32_16x16x32_f16(false, a1, false, bb, (short)0, acc1, false, false);
  }
  int col = col0 + (lane & 15);
  int rb = (lane >> 4) * 8;
  float bv = bias[col];
#pragma unroll
  for (int r = 0; r < 8; ++r) {
    H[(size_t)(row0 + rb + r) * FEAT + col] = acc0[r] + bv;
    H[(size_t)(row0 + 16 + rb + r) * FEAT + col] = acc1[r] + bv;
  }
}

// 5) BN stats: per-256-row-chunk partial sums/sumsq per feature (deterministic)
__global__ __launch_bounds__(256) void k_bnpart(const float* __restrict__ H, float* __restrict__ part) {
  int t = threadIdx.x;
  size_t r0 = (size_t)blockIdx.x * 256;
  float s0 = 0, s1 = 0, q0 = 0, q1 = 0;
  for (int i = 0; i < 256; ++i) {
    const float* row = H + (r0 + i) * FEAT;
    float a = row[t], b = row[t + 256];
    s0 += a; q0 += a * a;
    s1 += b; q1 += b * b;
  }
  part[(size_t)blockIdx.x * FEAT + t] = s0;
  part[(size_t)blockIdx.x * FEAT + t + 256] = s1;
  part[(size_t)BNBLK * FEAT + (size_t)blockIdx.x * FEAT + t] = q0;
  part[(size_t)BNBLK * FEAT + (size_t)blockIdx.x * FEAT + t + 256] = q1;
}

// 6) BN stats final: mean + invstd per feature
__global__ __launch_bounds__(512) void k_bnfinal(const float* __restrict__ part, float* __restrict__ stats) {
  int j = threadIdx.x;
  float s = 0, q = 0;
  for (int b = 0; b < BNBLK; ++b) {
    s += part[(size_t)b * FEAT + j];
    q += part[(size_t)BNBLK * FEAT + (size_t)b * FEAT + j];
  }
  float mean = s / (float)N_PIX;
  float var = q / (float)N_PIX - mean * mean;
  stats[j] = mean;
  stats[FEAT + j] = rsqrtf(var + 1e-5f);
}

// 7) per-row BN + ReLU + LayerNorm + l2n -> c (f32 in place over H) and c16
__global__ __launch_bounds__(256) void k_rownorm(float* __restrict__ Hc, _Float16* __restrict__ C16,
                                                 const float* __restrict__ stats, const float* __restrict__ bn_g,
                                                 const float* __restrict__ bn_b, const float* __restrict__ fn_w,
                                                 const float* __restrict__ fn_b) {
  __shared__ float red[256];
  size_t row = blockIdx.x;
  int t = threadIdx.x;
  float x0 = Hc[row * FEAT + t], x1 = Hc[row * FEAT + t + 256];
  x0 = fmaxf((x0 - stats[t]) * stats[FEAT + t] * bn_g[t] + bn_b[t], 0.0f);
  x1 = fmaxf((x1 - stats[t + 256]) * stats[FEAT + t + 256] * bn_g[t + 256] + bn_b[t + 256], 0.0f);
  float mu = breduce<256>(x0 + x1, red) / (float)FEAT;
  float d0 = x0 - mu, d1 = x1 - mu;
  float var = breduce<256>(d0 * d0 + d1 * d1, red) / (float)FEAT;
  float rs = rsqrtf(var + 1e-5f);
  float y0 = d0 * rs * fn_w[t] + fn_b[t];
  float y1 = d1 * rs * fn_w[t + 256] + fn_b[t + 256];
  float nrm = sqrtf(breduce<256>(y0 * y0 + y1 * y1, red)) + 1e-10f;
  y0 /= nrm; y1 /= nrm;
  Hc[row * FEAT + t] = y0;
  Hc[row * FEAT + t + 256] = y1;
  C16[row * FEAT + t] = (_Float16)y0;
  C16[row * FEAT + t + 256] = (_Float16)y1;
}

// 8) GEMM2: logits[N,190] = c16 @ Pn16^T, fused max/LN/argmax epilogue.
//    Block = 32 rows x 192 cols (8 waves: wm in {0,1} x wn in {0..3} x 3 col tiles).
__global__ __launch_bounds__(256) void k_gemm2(const _Float16* __restrict__ Cm, const _Float16* __restrict__ P,
                                               const int* __restrict__ gt, const float* __restrict__ mn_w,
                                               const float* __restrict__ mn_b, float* __restrict__ out_seg,
                                               float* __restrict__ out_log, int* __restrict__ correct) {
  __shared__ float lg[32][192];
  __shared__ float seg[32][19];
  int w = threadIdx.x >> 5;
  int lane = threadIdx.x & 31;
  int wm = w >> 2, wn = w & 3;
  int rowb = blockIdx.x * 32;
  int row0 = rowb + wm * 16;
  v8f zero = {};
  v8f acc[3];
#pragma unroll
  for (int j = 0; j < 3; ++j) acc[j] = zero;
  for (int kk = 0; kk < FEAT; kk += 32) {
    v16h a = load_fragA(Cm + (size_t)row0 * FEAT + kk, FEAT);
#pragma unroll
    for (int j = 0; j < 3; ++j) {
      int col0 = (wn + j * 4) * 16;
      v16h bb = load_fragB(P + (size_t)col0 * FEAT + kk, FEAT);
      acc[j] = __builtin_amdgcn_wmma_f32_16x16x32_f16(false, a, false, bb, (short)0, acc[j], false, false);
    }
  }
  int rb = (lane >> 4) * 8;
  int cl = lane & 15;
#pragma unroll
  for (int j = 0; j < 3; ++j) {
    int col = (wn + j * 4) * 16 + cl;
#pragma unroll
    for (int r = 0; r < 8; ++r) {
      int ri = wm * 16 + rb + r;
      float v = acc[j][r];
      lg[ri][col] = v;
      if (col < KM) out_log[(size_t)(rowb + ri) * KM + col] = v;
    }
  }
  __syncthreads();
  for (int idx = threadIdx.x; idx < 32 * K_CLS; idx += 256) {
    int i = idx / K_CLS, k = idx % K_CLS;
    float mx = lg[i][k * M_PRO];
#pragma unroll
    for (int m = 1; m < M_PRO; ++m) mx = fmaxf(mx, lg[i][k * M_PRO + m]);
    seg[i][k] = mx;
  }
  __syncthreads();
  if (threadIdx.x < 32) {
    int i = threadIdx.x;
    int n = rowb + i;
    float s = 0, s2 = 0;
    for (int k = 0; k < K_CLS; ++k) { float v = seg[i][k]; s += v; s2 += v * v; }
    float mu = s / (float)K_CLS;
    float var = s2 / (float)K_CLS - mu * mu;
    float rs = rsqrtf(var + 1e-5f);
    int b = n / HWP, p = n % HWP;
    float best = -1e30f; int bi = 0;
    for (int k = 0; k < K_CLS; ++k) {
      float v = (seg[i][k] - mu) * rs * mn_w[k] + mn_b[k];
      out_seg[(size_t)(b * K_CLS + k) * HWP + p] = v;
      if (v > best) { best = v; bi = k; }
    }
    correct[n] = (gt[n] == bi) ? 1 : 0;
  }
}

// 9) Per-class distributed sinkhorn (3 iters) + argmax -> proto_target, Q scaled by Bconst
__global__ __launch_bounds__(512) void k_sinkhorn(const float* __restrict__ logits, const int* __restrict__ gt,
                                                  float* __restrict__ Q, float* __restrict__ tgt) {
  __shared__ float red[512];
  int k = blockIdx.x;
  float* Qk = Q + (size_t)k * N_PIX * M_PRO;
  // pass A: count + total
  float cnt = 0.f, tot = 0.f;
  for (int n = threadIdx.x; n < N_PIX; n += 512) {
    if (gt[n] == k) {
      cnt += 1.0f;
      const float* L = logits + (size_t)n * KM + k * M_PRO;
#pragma unroll
      for (int m = 0; m < M_PRO; ++m) tot += L[m];
    }
  }
  cnt = breduce<512>(cnt, red);
  tot = breduce<512>(tot, red);
  float Bc = cnt * (float)M_PRO;
  float s0 = 1.0f / safef(tot);
  // pass B: init Q
  for (int n = threadIdx.x; n < N_PIX; n += 512) {
    bool mem = (gt[n] == k);
    const float* L = logits + (size_t)n * KM + k * M_PRO;
    float* q = Qk + (size_t)n * M_PRO;
#pragma unroll
    for (int m = 0; m < M_PRO; ++m) q[m] = mem ? L[m] * s0 : 0.0f;
  }
  __threadfence();
  __syncthreads();
  // 3 sinkhorn iterations
  for (int it = 0; it < 3; ++it) {
    float loc[M_PRO];
#pragma unroll
    for (int m = 0; m < M_PRO; ++m) loc[m] = 0.0f;
    for (int n = threadIdx.x; n < N_PIX; n += 512) {
      const float* q = Qk + (size_t)n * M_PRO;
#pragma unroll
      for (int m = 0; m < M_PRO; ++m) loc[m] += q[m];
    }
    float csum[M_PRO];
#pragma unroll
    for (int m = 0; m < M_PRO; ++m) csum[m] = safef(breduce<512>(loc[m], red));
    float sB = safef(Bc);
    for (int n = threadIdx.x; n < N_PIX; n += 512) {
      float* q = Qk + (size_t)n * M_PRO;
      float v[M_PRO];
      float rs = 0.0f;
#pragma unroll
      for (int m = 0; m < M_PRO; ++m) { v[m] = q[m] / csum[m] / (float)M_PRO; rs += v[m]; }
      rs = safef(rs);
#pragma unroll
      for (int m = 0; m < M_PRO; ++m) q[m] = v[m] / rs / sB;
    }
    __threadfence();
    __syncthreads();
  }
  // final: *= Bconst, argmax -> proto_target for members
  for (int n = threadIdx.x; n < N_PIX; n += 512) {
    float* q = Qk + (size_t)n * M_PRO;
    float best = -1e30f; int bi = 0;
#pragma unroll
    for (int m = 0; m < M_PRO; ++m) {
      float v = q[m] * Bc;
      q[m] = v;
      if (v > best) { best = v; bi = m; }
    }
    if (gt[n] == k) tgt[n] = (float)(k * M_PRO + bi);
  }
}

// 10) f partials: fpart[k][chunk][m][d] = sum over chunk rows (member & correct) Q*c
__global__ __launch_bounds__(512) void k_fpart(const float* __restrict__ Q, const float* __restrict__ Cf,
                                               const int* __restrict__ gt, const int* __restrict__ correct,
                                               float* __restrict__ fpart) {
  int k = blockIdx.y;
  int ch = blockIdx.x;
  int d = threadIdx.x;
  const float* Qk = Q + (size_t)k * N_PIX * M_PRO;
  float acc[M_PRO];
#pragma unroll
  for (int m = 0; m < M_PRO; ++m) acc[m] = 0.0f;
  int n0 = ch * 2048;
  for (int n = n0; n < n0 + 2048; ++n) {
    if (gt[n] == k && correct[n]) {
      const float* q = Qk + (size_t)n * M_PRO;
      float cv = Cf[(size_t)n * FEAT + d];
#pragma unroll
      for (int m = 0; m < M_PRO; ++m) acc[m] += q[m] * cv;
    }
  }
  float* fp = fpart + (((size_t)k * NCHUNK + ch) * M_PRO) * FEAT;
#pragma unroll
  for (int m = 0; m < M_PRO; ++m) fp[(size_t)m * FEAT + d] = acc[m];
}

// 11) per-class finalize: reduce fpart, n_m, EMA update, l2n -> d_out
__global__ __launch_bounds__(256) void k_finalize(const float* __restrict__ fpart, const float* __restrict__ Q,
                                                  const int* __restrict__ gt, const int* __restrict__ correct,
                                                  const float* __restrict__ pn32, float* __restrict__ outp) {
  __shared__ float fs[M_PRO * FEAT];
  __shared__ float red[256];
  __shared__ float nmsh[M_PRO];
  int k = blockIdx.x;
  for (int idx = threadIdx.x; idx < M_PRO * FEAT; idx += 256) {
    float s = 0.0f;
    for (int ch = 0; ch < NCHUNK; ++ch)
      s += fpart[(((size_t)k * NCHUNK + ch) * M_PRO) * FEAT + idx];
    fs[idx] = s;
  }
  float nm[M_PRO];
#pragma unroll
  for (int m = 0; m < M_PRO; ++m) nm[m] = 0.0f;
  for (int n = threadIdx.x; n < N_PIX; n += 256) {
    if (gt[n] == k && correct[n]) {
      const float* q = Q + (size_t)k * N_PIX * M_PRO + (size_t)n * M_PRO;
#pragma unroll
      for (int m = 0; m < M_PRO; ++m) nm[m] += q[m];
    }
  }
  __syncthreads();
  for (int m = 0; m < M_PRO; ++m) {
    float v = breduce<256>(nm[m], red);
    if (threadIdx.x == 0) nmsh[m] = v;
  }
  __syncthreads();
  float nsum = 0.0f;
#pragma unroll
  for (int m = 0; m < M_PRO; ++m) nsum += nmsh[m];
  float fnorm[M_PRO];
  for (int m = 0; m < M_PRO; ++m) {
    float a = fs[m * FEAT + threadIdx.x];
    float b = fs[m * FEAT + 256 + threadIdx.x];
    fnorm[m] = sqrtf(breduce<256>(a * a + b * b, red));
  }
  for (int idx = threadIdx.x; idx < M_PRO * FEAT; idx += 256) {
    int m = idx >> 9;
    float fn = fs[idx] / fmaxf(fnorm[m], 1e-12f);
    bool upd = (nmsh[m] != 0.0f) && (nsum > 0.0f);
    float old = pn32[(size_t)(k * M_PRO) * FEAT + idx];
    fs[idx] = upd ? 0.999f * old + 0.001f * fn : old;
  }
  __syncthreads();
  float nn[M_PRO];
  for (int m = 0; m < M_PRO; ++m) {
    float a = fs[m * FEAT + threadIdx.x];
    float b = fs[m * FEAT + 256 + threadIdx.x];
    nn[m] = sqrtf(breduce<256>(a * a + b * b, red)) + 1e-10f;
  }
  for (int idx = threadIdx.x; idx < M_PRO * FEAT; idx += 256) {
    int m = idx >> 9;
    outp[(size_t)(k * M_PRO) * FEAT + idx] = fs[idx] / nn[m];
  }
}

// ---------------------------------------------------------------------------
extern "C" void kernel_launch(void* const* d_in, const int* in_sizes, int n_in,
                              void* d_out, int out_size, void* d_ws, size_t ws_size,
                              hipStream_t stream) {
  const float* features = (const float*)d_in[0];
  const int*   gt       = (const int*)d_in[1];
  const float* proj_w   = (const float*)d_in[2];
  const float* proj_b   = (const float*)d_in[3];
  const float* bn_g     = (const float*)d_in[4];
  const float* bn_b     = (const float*)d_in[5];
  const float* fn_w     = (const float*)d_in[6];
  const float* fn_b     = (const float*)d_in[7];
  const float* mn_w     = (const float*)d_in[8];
  const float* mn_b     = (const float*)d_in[9];
  const float* protos   = (const float*)d_in[10];

  float* out = (float*)d_out;
  float* out_seg = out + OUT_SEG_OFF;
  float* out_log = out + OUT_LOG_OFF;
  float* out_tgt = out + OUT_TGT_OFF;
  float* out_pro = out + OUT_PRO_OFF;

  char* ws = (char*)d_ws;
  constexpr size_t SZ_FLAT16 = (size_t)N_PIX * FEAT * 2;        // also reused as c16
  constexpr size_t SZ_H      = (size_t)N_PIX * FEAT * 4;        // h, then c (in place)
  constexpr size_t SZ_W16    = (size_t)FEAT * FEAT * 2;
  constexpr size_t SZ_PN32   = (size_t)KM * FEAT * 4;
  constexpr size_t SZ_PN16   = (size_t)KM_PAD * FEAT * 2;
  constexpr size_t SZ_BNP    = (size_t)2 * BNBLK * FEAT * 4;
  constexpr size_t SZ_STATS  = (size_t)2 * FEAT * 4;
  constexpr size_t SZ_CORR   = (size_t)N_PIX * 4;
  constexpr size_t SZ_Q      = (size_t)K_CLS * N_PIX * M_PRO * 4;

  size_t o = 0;
  _Float16* w_flat16 = (_Float16*)(ws + o); o += SZ_FLAT16;
  float*    w_h      = (float*)(ws + o);    o += SZ_H;
  _Float16* w_w16    = (_Float16*)(ws + o); o += SZ_W16;
  float*    w_pn32   = (float*)(ws + o);    o += SZ_PN32;
  _Float16* w_pn16   = (_Float16*)(ws + o); o += SZ_PN16;
  float*    w_bnp    = (float*)(ws + o);    o += SZ_BNP;
  float*    w_stats  = (float*)(ws + o);    o += SZ_STATS;
  int*      w_corr   = (int*)(ws + o);      o += SZ_CORR;
  float*    w_Q      = (float*)(ws + o);    o += SZ_Q;
  float*    w_fpart  = (float*)(ws + o);    // 19*36*10*512*4

  _Float16* w_c16 = w_flat16;  // reuse after GEMM1 consumes flat16
  float*    w_c32 = w_h;       // c overwrites h in place

  k_convw<<<dim3((FEAT * FEAT + 255) / 256), dim3(256), 0, stream>>>(proj_w, w_w16);
  k_protos<<<dim3(KM_PAD), dim3(256), 0, stream>>>(protos, w_pn32, w_pn16);
  k_transpose<<<dim3(HWP / 32, FEAT / 32, 8), dim3(32, 8), 0, stream>>>(features, w_flat16);
  k_gemm1<<<dim3(FEAT / 64, N_PIX / 64), dim3(256), 0, stream>>>(w_flat16, w_w16, proj_b, w_h);
  k_bnpart<<<dim3(BNBLK), dim3(256), 0, stream>>>(w_h, w_bnp);
  k_bnfinal<<<dim3(1), dim3(512), 0, stream>>>(w_bnp, w_stats);
  k_rownorm<<<dim3(N_PIX), dim3(256), 0, stream>>>(w_h, w_c16, w_stats, bn_g, bn_b, fn_w, fn_b);
  k_gemm2<<<dim3(N_PIX / 32), dim3(256), 0, stream>>>(w_c16, w_pn16, gt, mn_w, mn_b, out_seg, out_log, w_corr);
  k_sinkhorn<<<dim3(K_CLS), dim3(512), 0, stream>>>(out_log, gt, w_Q, out_tgt);
  k_fpart<<<dim3(NCHUNK, K_CLS), dim3(512), 0, stream>>>(w_Q, w_c32, gt, w_corr, w_fpart);
  k_finalize<<<dim3(K_CLS), dim3(256), 0, stream>>>(w_fpart, w_Q, gt, w_corr, w_pn32, out_pro);
}
